// MaskPropagation_35974646071558
// MI455X (gfx1250) — matrix operand order repaired and checked by reading, not compiled, not run on hardware
//
#include <hip/hip_runtime.h>

// ---- problem constants (from reference setup_inputs) ----
#define RAD   6
#define P     13
#define PP    169            // P*P
#define TOPK  36
#define CCH   256            // channels
#define H     120
#define W     216
#define NPIX  (H * W)
#define OBJ   11
#define H4    (H * 4)
#define W4    (W * 4)
#define CSTRIDE 347          // padded corr row stride (dwords), odd -> conflict-free scans
#define BCOLSTR 264          // staged B column stride in bf16 units (528B): 16 lanes hit 16 banks

#if __has_builtin(__builtin_amdgcn_global_load_async_to_lds_b128) && \
    __has_builtin(__builtin_amdgcn_s_wait_asynccnt)
#define HAVE_ASYNC_LDS 1
#else
#define HAVE_ASYNC_LDS 0
#endif

typedef __attribute__((ext_vector_type(16))) __bf16 v16bf;
typedef __attribute__((ext_vector_type(8)))  __bf16 v8bf;
typedef __attribute__((ext_vector_type(8)))  float  v8f;

// exact parameter types of the async builtin (from hipcc diagnostic):
// arg0: addrspace(1) pointer to GCC-style int4 vector; arg1: LDS equivalent
typedef int v4i_g __attribute__((vector_size(16)));
typedef __attribute__((address_space(1))) v4i_g* gp_v4i;
typedef __attribute__((address_space(3))) v4i_g* lp_v4i;

union U16 { v16bf v; v8bf h[2]; unsigned u[8]; };
union U8  { v8bf  v; unsigned u[4]; };

__device__ __forceinline__ unsigned pack_bf16x2(float a, float b) {
    unsigned ua = __builtin_bit_cast(unsigned, a);
    unsigned ub = __builtin_bit_cast(unsigned, b);
    ua += 0x7FFFu + ((ua >> 16) & 1u);           // RNE
    ub += 0x7FFFu + ((ub >> 16) & 1u);
    return (ua >> 16) | (ub & 0xFFFF0000u);
}

// 16B global -> LDS copy; async DMA (ASYNCcnt) when the toolchain has it.
__device__ __forceinline__ void copy16_g2l(const void* g, void* l) {
#if HAVE_ASYNC_LDS
    // generic LDS pointer keeps the LDS offset in its low 32 bits (ISA aperture rule)
    __builtin_amdgcn_global_load_async_to_lds_b128(
        (gp_v4i)(size_t)g,
        (lp_v4i)(unsigned)(size_t)l,
        0, 0);
#else
    *(v8bf*)l = *(const v8bf*)g;
#endif
}

// =====================================================================
// Prep: [c][h][w] f32 -> [pixel][c] bf16 (three tensors, grid.y = 0..2).
// Query features pre-scaled by 1/sqrt(256) = 2^-4 (exact in bf16).
// =====================================================================
__global__ __launch_bounds__(256)
void prep_kernel(const float* __restrict__ fq,
                 const float* __restrict__ fm0,
                 const float* __restrict__ fm1,
                 __bf16* __restrict__ ws)
{
    const int t = blockIdx.y;
    const float* src = (t == 0) ? fq : (t == 1) ? fm0 : fm1;
    const float sc = (t == 0) ? 0.0625f : 1.0f;
    unsigned* dst = (unsigned*)(ws + (size_t)t * NPIX * CCH);

    int idx = blockIdx.x * 256 + threadIdx.x;     // 0 .. NPIX*16-1
    int g = idx / NPIX;                           // 16-channel group
    int p = idx - g * NPIX;                       // pixel (coalesced across lanes)

    const float* s = src + (size_t)g * 16 * NPIX + p;
    unsigned u[8];
    #pragma unroll
    for (int i = 0; i < 8; ++i)
        u[i] = pack_bf16x2(s[(2 * i) * NPIX] * sc, s[(2 * i + 1) * NPIX] * sc);

    unsigned* d = dst + (size_t)p * 128 + g * 8;
    uint4 a; a.x = u[0]; a.y = u[1]; a.z = u[2]; a.w = u[3];
    uint4 b; b.x = u[4]; b.y = u[5]; b.z = u[6]; b.w = u[7];
    *(uint4*)(d)     = a;
    *(uint4*)(d + 4) = b;
}

// =====================================================================
// Main fused kernel: 4 waves/block = 4 consecutive query rows.
// Memory rows staged once into LDS (bf16), shared by all waves.
// =====================================================================
template<bool PRE>
__global__ __launch_bounds__(128)
void maskprop_main(const float* __restrict__ fqf,
                   const float* __restrict__ fm0f,
                   const float* __restrict__ fm1f,
                   const __bf16* __restrict__ qbf,
                   const __bf16* __restrict__ m0bf,
                   const __bf16* __restrict__ m1bf,
                   const float* __restrict__ mk0,
                   const float* __restrict__ mk1,
                   float* __restrict__ out)
{
    extern __shared__ char smem[];
    float*  corrAll = (float*)smem;                         // 4 * 16 * CSTRIDE
    float*  gmaxAll = corrAll + 4 * 16 * CSTRIDE;           // 4 * 16 * 26
    __bf16* Bbuf    = (__bf16*)(gmaxAll + 4 * 16 * 26);     // 32 * BCOLSTR
    float*  dumpA   = (float*)(Bbuf + 32 * BCOLSTR);        // 128 dwords (dead-store sink)

    const int tid  = threadIdx.x;
    const int wv   = tid >> 5;                 // wave id = query-row offset
    const int lane = tid & 31;
    const int n    = lane & 15;                // B/C column; query id in phase 2
    const int half = lane >> 4;
    const int x0   = blockIdx.x * 16;
    const int y0   = blockIdx.y * 4;
    const int y    = y0 + wv;                  // this wave's query row

    float* corrW  = corrAll + wv * 16 * CSTRIDE;
    float* gmaxW  = gmaxAll + wv * 16 * 26;
    float* myDump = dumpA + tid;

    // staging geometry for this thread (constant for whole kernel)
    const int  scol = tid >> 2;                // 0..31 staged column
    const int  sq   = tid & 3;                 // 64-channel quarter
    const int  smx  = x0 - RAD + scol;         // source x of my column
    const bool sok  = (smx >= 0) && (smx < W);
    __bf16*    sdst = Bbuf + scol * BCOLSTR + sq * 64;

    // zero OOB column slices ONCE (staging never touches them again;
    // disjoint from valid-column writes, visible after the first barrier)
    if (!sok) {
        U8 z; z.u[0] = z.u[1] = z.u[2] = z.u[3] = 0;
        #pragma unroll
        for (int i = 0; i < 8; ++i) *(v8bf*)(sdst + i * 8) = z.v;
    }

    // ---------------- A matrix: 16 queries x 256 ch, bf16 WMMA layout ----------------
    // lanes 0-15: row M=lane, K elems {0..7,16..23}; lanes 16-31: row M=lane-16, K {8..15,24..31}
    v16bf A[8];
    {
        int xq = x0 + n; if (xq >= W) xq = W - 1;     // clamp; garbage rows never stored
        if (PRE) {
            const __bf16* qb = qbf + (size_t)(y * W + xq) * CCH;
            #pragma unroll
            for (int kc = 0; kc < 8; ++kc) {
                const int c1 = kc * 32 + half * 8;
                U16 a;
                a.h[0] = *(const v8bf*)(qb + c1);          // K run 1 (8 contiguous)
                a.h[1] = *(const v8bf*)(qb + c1 + 16);     // K run 2
                A[kc] = a.v;
            }
        } else {
            const float* qb = fqf + y * W + xq;
            #pragma unroll
            for (int kc = 0; kc < 8; ++kc) {
                U16 a;
                #pragma unroll
                for (int e = 0; e < 8; ++e) {
                    int c1 = kc * 32 + half * 8 + e;       // run 1
                    int c2 = kc * 32 + 16 + half * 8 + e;  // run 2
                    float f1 = qb[c1 * NPIX] * 0.0625f;    // fold 1/sqrt(c)
                    float f2 = qb[c2 * NPIX] * 0.0625f;
                    if (e & 1) { a.u[(e >> 1)]     |= pack_bf16x2(0.f, f1) & 0xFFFF0000u;
                                 a.u[(e >> 1) + 4] |= pack_bf16x2(0.f, f2) & 0xFFFF0000u; }
                    else       { a.u[(e >> 1)]      = pack_bf16x2(f1, 0.f) & 0x0000FFFFu;
                                 a.u[(e >> 1) + 4]  = pack_bf16x2(f2, 0.f) & 0x0000FFFFu; }
                }
                A[kc] = a.v;
            }
        }
    }

    // ---------------- Phase 1: stage memory rows, WMMA, band-extract ----------------
    #pragma unroll 1
    for (int fr = 0; fr < 2; ++fr) {
        const __bf16* fmb = fr ? m1bf : m0bf;
        const float*  fmf = fr ? fm1f : fm0f;
        #pragma unroll 1
        for (int ridx = 0; ridx < 16; ++ridx) {
            const int  my       = y0 - RAD + ridx;        // memory row for this step
            const bool rowvalid = (my >= 0) && (my < H);  // block-uniform

            __syncthreads();                              // prior consumers done
            if (rowvalid && sok) {                        // stage my 128B slice
                if (PRE) {
                    const __bf16* gs = fmb + (size_t)(my * W + smx) * CCH + sq * 64;
                    #pragma unroll
                    for (int i = 0; i < 8; ++i)
                        copy16_g2l(gs + i * 8, sdst + i * 8);
                } else {
                    const float* gs = fmf + my * W + smx;
                    #pragma unroll
                    for (int i = 0; i < 8; ++i) {
                        U8 d;
                        #pragma unroll
                        for (int k2 = 0; k2 < 4; ++k2) {
                            int ch = sq * 64 + i * 8 + 2 * k2;
                            d.u[k2] = pack_bf16x2(gs[ch * NPIX], gs[(ch + 1) * NPIX]);
                        }
                        *(v8bf*)(sdst + i * 8) = d.v;
                    }
                }
            }
            // prefetch next round's slice while this round computes
            if (PRE && sok && ridx < 15) {
                const int myn = my + 1;
                if (myn >= 0 && myn < H)
                    __builtin_prefetch(fmb + (size_t)(myn * W + smx) * CCH + sq * 64, 0, 3);
            }
#if HAVE_ASYNC_LDS
            if (PRE) __builtin_amdgcn_s_wait_asynccnt(0); // barrier doesn't track ASYNCcnt
#endif
            __syncthreads();                              // staging visible

            const int di = ridx - wv;                     // wave-uniform
            if (di >= 0 && di < P) {
                v8f acc0 = {}; v8f acc1 = {};
                if (rowvalid) {
                    const __bf16* c0 = Bbuf + n * BCOLSTR;          // tile 0 column
                    const __bf16* c1 = Bbuf + (n + 16) * BCOLSTR;   // tile 1 column
                    U16 b0, b1, nb0, nb1;
                    b0.h[0] = *(const v8bf*)(c0 + half * 8);
                    b0.h[1] = *(const v8bf*)(c0 + half * 8 + 16);
                    b1.h[0] = *(const v8bf*)(c1 + half * 8);
                    b1.h[1] = *(const v8bf*)(c1 + half * 8 + 16);
                    #pragma unroll
                    for (int kc = 0; kc < 8; ++kc) {
                        if (kc < 7) {
                            const int c = (kc + 1) * 32 + half * 8;
                            nb0.h[0] = *(const v8bf*)(c0 + c);
                            nb0.h[1] = *(const v8bf*)(c0 + c + 16);
                            nb1.h[0] = *(const v8bf*)(c1 + c);
                            nb1.h[1] = *(const v8bf*)(c1 + c + 16);
                        }
                        acc0 = __builtin_amdgcn_wmma_f32_16x16x32_bf16(
                                   false, A[kc], false, b0.v, (short)0, acc0, false, false);
                        acc1 = __builtin_amdgcn_wmma_f32_16x16x32_bf16(
                                   false, A[kc], false, b1.v, (short)0, acc1, false, false);
                        b0 = nb0; b1 = nb1;
                    }
                }
                // Branchless band extraction: C VGPR r holds row M = r + 8*half,
                // lane holds col n. Exactly one of {tile0: dj0=n-M, tile1: dj0+16}
                // can be valid; invalid lanes store to a per-lane dump slot.
                const int cbase = fr * PP + di * P;
                #pragma unroll
                for (int r = 0; r < 8; ++r) {
                    const int  M    = r + half * 8;
                    const int  d0   = n - M;
                    const bool use0 = (d0 >= 0);
                    const int  dj   = use0 ? d0 : d0 + 16;      // >= 0 always
                    const float val = use0 ? acc0[r] : acc1[r];
                    float* p = (dj < P) ? (corrW + M * CSTRIDE + cbase + dj) : myDump;
                    *p = val;                                    // unconditional ds_store
                }
            }
        }
    }

    __syncthreads();

    // ====== Phase 2: fused top-36 + softmax + mask gather (lane pair / query) ======
    const int   j    = n;
    const int   xq   = x0 + j;
    const int   base = half * PP;                 // this lane owns frame `half`
    float*      row  = &corrW[j * CSTRIDE];
    float*      gm   = &gmaxW[j * 26 + half * P];

    #pragma unroll 1
    for (int g = 0; g < P; ++g) {                 // group maxima over dj per di
        float m = -INFINITY;
        #pragma unroll
        for (int t = 0; t < P; ++t) m = fmaxf(m, row[base + g * P + t]);
        gm[g] = m;
    }

    float outacc[6] = {0.f, 0.f, 0.f, 0.f, 0.f, 0.f};
    const int nobj  = half ? (OBJ - 6) : 6;       // split 11 channels 6/5 across pair
    const int obase = half ? 6 : 0;

    float v0 = 0.0f, denom = 0.0f;

    #pragma unroll 1
    for (int k = 0; k < TOPK; ++k) {
        float bm = -INFINITY; int bg = 0;
        #pragma unroll
        for (int g = 0; g < P; ++g) {
            float v = gm[g];
            if (v > bm) { bm = v; bg = g; }
        }
        int bt = 0;
        #pragma unroll
        for (int t = P - 1; t >= 0; --t)
            if (row[base + bg * P + t] == bm) bt = t;  // lowest matching index
        int mi = base + bg * P + bt;

        float om  = __shfl_xor(bm, 16, 32);
        int   omi = __shfl_xor(mi, 16, 32);
        const bool take_other = (om > bm) || (om == bm && omi < mi);
        const float mc  = take_other ? om : bm;
        const int   mic = take_other ? omi : mi;
        if (!take_other) {                        // I own the winner: remove + refresh
            row[mi] = -INFINITY;
            float m2 = -INFINITY;
            #pragma unroll
            for (int t = 0; t < P; ++t) m2 = fmaxf(m2, row[base + bg * P + t]);
            gm[bg] = m2;
        }

        if (k == 0) v0 = mc;
        const float e = __expf(mc - v0);
        denom += e;

        const int fr2 = (mic >= PP) ? 1 : 0;
        const int rr  = mic - fr2 * PP;
        const int di  = rr / P;
        const int dj  = rr - di * P;
        const int ym  = y + di - RAD;
        const int xm  = xq + dj - RAD;
        const float* mkp = fr2 ? mk1 : mk0;
        if (ym >= 0 && ym < H && xm >= 0 && xm < W && xq < W) {
            const float* mp = mkp + (4 * ym) * W4 + 4 * xm;   // ::4 strided mask read
            #pragma unroll
            for (int o = 0; o < 6; ++o)
                if (o < nobj) outacc[o] += e * mp[(obase + o) * (H4 * W4)];
        }
    }

    if (xq < W) {
        const float inv = 1.0f / denom;
        #pragma unroll
        for (int o = 0; o < 6; ++o)
            if (o < nobj) out[(obase + o) * NPIX + y * W + xq] = outacc[o] * inv;
    }
}

extern "C" void kernel_launch(void* const* d_in, const int* in_sizes, int n_in,
                              void* d_out, int out_size, void* d_ws, size_t ws_size,
                              hipStream_t stream) {
    (void)in_sizes; (void)n_in; (void)out_size;
    const float* fq  = (const float*)d_in[0];
    const float* fm0 = (const float*)d_in[1];
    const float* fm1 = (const float*)d_in[2];
    const float* mk0 = (const float*)d_in[3];
    const float* mk1 = (const float*)d_in[4];
    float* o = (float*)d_out;

    const size_t smemB = (size_t)(4 * 16 * CSTRIDE + 4 * 16 * 26) * 4
                       + (size_t)32 * BCOLSTR * 2
                       + 128 * 4;                 // dump area
    dim3 grid((W + 15) / 16, H / 4);             // 14 x 30 blocks, 4 waves each

    const size_t need = (size_t)3 * NPIX * CCH * sizeof(__bf16);   // 39.8 MB
    if (ws_size >= need) {
        __bf16* wsb = (__bf16*)d_ws;
        prep_kernel<<<dim3((NPIX * 16) / 256, 3), 256, 0, stream>>>(fq, fm0, fm1, wsb);
        maskprop_main<true><<<grid, 128, smemB, stream>>>(
            fq, fm0, fm1, wsb, wsb + (size_t)NPIX * CCH, wsb + (size_t)2 * NPIX * CCH,
            mk0, mk1, o);
    } else {
        maskprop_main<false><<<grid, 128, smemB, stream>>>(
            fq, fm0, fm1, nullptr, nullptr, nullptr, mk0, mk1, o);
    }
}
